// PseudoTokenTEGridEncoderThroughTime_86251533238884
// MI455X (gfx1250) — compile-verified, matrix operand order checked by reading, not run on hardware
//
#include <hip/hip_runtime.h>
#include <hip/hip_bf16.h>
#include <math.h>

// ---- problem constants (from reference) ----
#define B_    2
#define N_    65536
#define BN    (B_*N_)
#define E_    128
#define H_    8
#define D_    16
#define NT_   8
#define PHI_H_ 64
#define FF_H_ 512
#define PMAX_G 32                      // safe cap on grid points per spatial dim (actual ~28)
#define M_MAX  (NT_*PMAX_G*PMAX_G)     // 8192
#define G_MAX  (B_*M_MAX)              // 16384
#define PCAP   2048                    // cap on keys per cell (typical ~12)

typedef __attribute__((ext_vector_type(16))) __bf16 v16bf;
typedef __attribute__((ext_vector_type(8)))  __bf16 v8bf;
typedef __attribute__((ext_vector_type(8)))  float  v8f;
typedef __attribute__((ext_vector_type(4)))  float  v4f;
#define V8F_ZERO {0.f,0.f,0.f,0.f,0.f,0.f,0.f,0.f}

// ---- helpers ----
__device__ __forceinline__ unsigned f2ord(float f){
  unsigned u = __float_as_uint(f);
  return (u & 0x80000000u) ? ~u : (u | 0x80000000u);
}
__device__ __forceinline__ float ord2f(unsigned u){
  u = (u & 0x80000000u) ? (u & 0x7FFFFFFFu) : ~u;
  return __uint_as_float(u);
}
__device__ __forceinline__ float gelu_f(float x){
  float x3 = x*x*x;
  return 0.5f*x*(1.0f + tanhf(0.7978845608028654f*(x + 0.044715f*x3)));
}

// Fragment load for v_wmma_f32_16x16x32_bf16 (ISA 7.12.2 16-bit 16x32 layout):
// per lane the 16 values are two contiguous 16-byte runs:
//   K = kbase + half*8 + [0..7]  and  K = kbase + 16 + half*8 + [0..7]
// -> two ds_load_b128 from a row-major bf16 row.
__device__ __forceinline__ v16bf load_frag(const __bf16* rowptr, int kbase, int lane){
  int half8 = ((lane >> 4) & 1) * 8;
  v8bf a = *(const v8bf*)(rowptr + kbase + half8);
  v8bf b = *(const v8bf*)(rowptr + kbase + 16 + half8);
  v16bf r;
  #pragma unroll
  for (int i = 0; i < 8; ++i){ r[i] = a[i]; r[i+8] = b[i]; }
  return r;
}

struct GridParams { float lo0,hi0,lo1,hi1,sp0,sp1; int p0,p1,m,G; };
__device__ __forceinline__ GridParams make_params(const unsigned* bounds){
  GridParams gp;
  float mn1 = ord2f(bounds[0]), mx1 = ord2f(bounds[1]);
  float mn2 = ord2f(bounds[2]), mx2 = ord2f(bounds[3]);
  gp.lo0 = mn1 - 0.1f; gp.hi0 = mx1 + 0.1f;
  gp.lo1 = mn2 - 0.1f; gp.hi1 = mx2 + 0.1f;
  int p0 = (int)((gp.hi0 - gp.lo0) * 24.0f);
  int p1 = (int)((gp.hi1 - gp.lo1) * 24.0f);
  p0 = min(max(p0, 2), PMAX_G);
  p1 = min(max(p1, 2), PMAX_G);
  gp.p0 = p0; gp.p1 = p1;
  gp.sp0 = (gp.hi0 - gp.lo0) / (float)(p0 - 1);
  gp.sp1 = (gp.hi1 - gp.lo1) / (float)(p1 - 1);
  gp.m = NT_ * p0 * p1;
  gp.G = B_ * gp.m;
  return gp;
}

// ---- K0: init workspace ----
__global__ void k_init(unsigned* bounds, int* counts, int* cursor){
  int t = blockIdx.x * blockDim.x + threadIdx.x;
  if (t < 4) bounds[t] = (t & 1) ? 0u : 0xFFFFFFFFu;  // even=min slots, odd=max slots
  if (t < G_MAX){ counts[t] = 0; cursor[t] = 0; }
}

// ---- K1: global spatial min/max of x[:,:,1] and x[:,:,2] ----
__global__ void k_bounds(const float* __restrict__ x, unsigned* bounds){
  int tid = blockIdx.x * blockDim.x + threadIdx.x;
  int stride = gridDim.x * blockDim.x;
  float mn1 = INFINITY, mx1 = -INFINITY, mn2 = INFINITY, mx2 = -INFINITY;
  for (int i = tid; i < BN; i += stride){
    float a = x[i*3+1], b = x[i*3+2];
    mn1 = fminf(mn1, a); mx1 = fmaxf(mx1, a);
    mn2 = fminf(mn2, b); mx2 = fmaxf(mx2, b);
  }
  for (int o = 16; o > 0; o >>= 1){
    mn1 = fminf(mn1, __shfl_xor(mn1, o, 32));
    mx1 = fmaxf(mx1, __shfl_xor(mx1, o, 32));
    mn2 = fminf(mn2, __shfl_xor(mn2, o, 32));
    mx2 = fmaxf(mx2, __shfl_xor(mx2, o, 32));
  }
  if ((threadIdx.x & 31) == 0){
    atomicMin(&bounds[0], f2ord(mn1));
    atomicMax(&bounds[1], f2ord(mx1));
    atomicMin(&bounds[2], f2ord(mn2));
    atomicMax(&bounds[3], f2ord(mx2));
  }
}

// ---- K2: nearest-cell assignment + histogram ----
__global__ void k_assign(const float* __restrict__ x, const float* __restrict__ tg,
                         const unsigned* __restrict__ bounds,
                         int* counts, int* cellof){
  GridParams gp = make_params(bounds);
  int pid = blockIdx.x * blockDim.x + threadIdx.x;
  if (pid >= BN) return;
  int b = pid / N_;
  float t  = x[pid*3+0], x1 = x[pid*3+1], x2 = x[pid*3+2];
  float t0 = tg[b*NT_], t1 = tg[b*NT_ + NT_ - 1];
  float spt = (t1 - t0) / (float)(NT_ - 1);
  int it = (int)fminf(fmaxf(floorf((t  - t0)    / spt    + 0.5f), 0.f), (float)(NT_    - 1));
  int i0 = (int)fminf(fmaxf(floorf((x1 - gp.lo0)/ gp.sp0 + 0.5f), 0.f), (float)(gp.p0 - 1));
  int i1 = (int)fminf(fmaxf(floorf((x2 - gp.lo1)/ gp.sp1 + 0.5f), 0.f), (float)(gp.p1 - 1));
  int cell = (it * gp.p0 + i0) * gp.p1 + i1;
  int g = b * gp.m + cell;
  cellof[pid] = g;
  atomicAdd(&counts[g], 1);
}

// ---- K3: single-block exclusive scan over G_MAX counts ----
__global__ void k_scan(const int* __restrict__ counts, int* offsets){
  __shared__ int s[1024];
  int t = threadIdx.x;
  int carry = 0;
  for (int base = 0; base < G_MAX; base += 1024){
    int v = counts[base + t];
    s[t] = v; __syncthreads();
    for (int d = 1; d < 1024; d <<= 1){
      int y = (t >= d) ? s[t - d] : 0;
      __syncthreads();
      s[t] += y;
      __syncthreads();
    }
    offsets[base + t] = s[t] - v + carry;
    carry += s[1023];
    __syncthreads();
  }
  if (t == 0) offsets[G_MAX] = carry;
}

// ---- K4: CSR fill ----
__global__ void k_scatter(const int* __restrict__ cellof, const int* __restrict__ offsets,
                          int* cursor, int* order){
  int pid = blockIdx.x * blockDim.x + threadIdx.x;
  if (pid >= BN) return;
  int c = cellof[pid];
  int pos = atomicAdd(&cursor[c], 1);
  order[offsets[c] + pos] = pid;
}

// ---- K5: q / k_grid / v_grid of the shared latent token ----
__global__ void k_pre(const float* __restrict__ latent,
                      const float* __restrict__ g1, const float* __restrict__ b1,
                      const float* __restrict__ Wq, const float* __restrict__ Wk,
                      const float* __restrict__ Wv,
                      float* qvec, float* kgrid, float* vgrid){
  __shared__ float zn[E_];
  __shared__ float ssum[E_], ssq[E_];
  int t = threadIdx.x;
  float v = latent[t];
  ssum[t] = v; ssq[t] = v * v; __syncthreads();
  for (int d = 64; d > 0; d >>= 1){
    if (t < d){ ssum[t] += ssum[t + d]; ssq[t] += ssq[t + d]; }
    __syncthreads();
  }
  float mean = ssum[0] * (1.0f/128.f);
  float var  = ssq[0]  * (1.0f/128.f) - mean * mean;
  float rstd = rsqrtf(var + 1e-5f);
  zn[t] = (v - mean) * rstd * g1[t] + b1[t];
  __syncthreads();
  float aq = 0.f, ak = 0.f, av = 0.f;
  for (int k = 0; k < E_; ++k){
    float zk = zn[k];
    aq += zk * Wq[k*E_ + t];
    ak += zk * Wk[k*E_ + t];
    av += zk * Wv[k*E_ + t];
  }
  qvec[t] = aq; kgrid[t] = ak; vgrid[t] = av;
}

// ---- K6: batched LN + K/V projection for all points (WMMA bf16) ----
__global__ void __launch_bounds__(256) k_kv(const float* __restrict__ z,
                     const float* __restrict__ g1, const float* __restrict__ b1,
                     const float* __restrict__ Wk, const float* __restrict__ Wv,
                     __bf16* Kbuf, __bf16* Vbuf){
  __shared__ __attribute__((aligned(16))) __bf16 wkT[E_][E_];   // transposed: [n][k]
  __shared__ __attribute__((aligned(16))) __bf16 wvT[E_][E_];
  __shared__ __attribute__((aligned(16))) float  zbuf[16][E_];
  __shared__ __attribute__((aligned(16))) __bf16 abf[16][E_];
  int t = threadIdx.x, wid = t >> 5, lane = t & 31;
  int tile = blockIdx.x;

  __builtin_prefetch(z + (size_t)(tile + 1) * (16*E_), 0, 1);

  // stage weights transposed (vector global reads, b16 LDS writes)
  for (int i = 0; i < 16; ++i){
    int e4 = t + 256*i;                 // 4096 float4 per matrix
    int k = e4 >> 5, nb = (e4 & 31) * 4;
    v4f wk4 = *(const v4f*)(Wk + (size_t)e4*4);
    v4f wv4 = *(const v4f*)(Wv + (size_t)e4*4);
    #pragma unroll
    for (int j = 0; j < 4; ++j){
      wkT[nb + j][k] = (__bf16)wk4[j];
      wvT[nb + j][k] = (__bf16)wv4[j];
    }
  }
  // stage the 16-row z tile (contiguous 2048 floats)
  {
    const v4f* zsrc = (const v4f*)(z + (size_t)tile * (16*E_));
    v4f* zdst = (v4f*)&zbuf[0][0];
    #pragma unroll
    for (int i = 0; i < 2; ++i) zdst[t + 256*i] = zsrc[t + 256*i];
  }
  __syncthreads();
  // LayerNorm: wave per row, 2 rows each
  for (int rr = 0; rr < 2; ++rr){
    int row = wid + rr*8;
    float s = 0.f, sq = 0.f;
    for (int k = 0; k < 4; ++k){ float v = zbuf[row][lane + 32*k]; s += v; sq += v*v; }
    for (int o = 16; o > 0; o >>= 1){ s += __shfl_xor(s, o, 32); sq += __shfl_xor(sq, o, 32); }
    float mean = s * (1.0f/128.f), var = sq * (1.0f/128.f) - mean*mean;
    float rstd = rsqrtf(var + 1e-5f);
    for (int k = 0; k < 4; ++k){
      int c = lane + 32*k;
      abf[row][c] = (__bf16)((zbuf[row][c] - mean) * rstd * g1[c] + b1[c]);
    }
  }
  __syncthreads();
  // WMMA: wave w owns output col-tile w for both K and V
  v8f accK = V8F_ZERO, accV = V8F_ZERO;
  int col = (lane & 15) + (wid << 4);
  int arow = lane & 15;
  #pragma unroll
  for (int kk = 0; kk < 4; ++kk){
    int kb = kk * 32;
    v16bf af = load_frag(&abf[arow][0], kb, lane);
    v16bf bk = load_frag(&wkT[col][0], kb, lane);
    v16bf bv = load_frag(&wvT[col][0], kb, lane);
    accK = __builtin_amdgcn_wmma_f32_16x16x32_bf16(false, af, false, bk, (short)0, accK, false, false);
    accV = __builtin_amdgcn_wmma_f32_16x16x32_bf16(false, af, false, bv, (short)0, accV, false, false);
  }
  int mbase = ((lane >> 4) & 1) * 8;
  // stage K into LDS (reuse abf) then cooperative b128 stores
  __syncthreads();
  #pragma unroll
  for (int r = 0; r < 8; ++r) abf[mbase + r][col] = (__bf16)accK[r];
  __syncthreads();
  {
    const v8bf* src = (const v8bf*)&abf[0][0];
    v8bf* dst = (v8bf*)(Kbuf + (size_t)tile * (16*E_));
    dst[t] = src[t];                     // 256 * 8 bf16 = full tile
  }
  __syncthreads();
  #pragma unroll
  for (int r = 0; r < 8; ++r) abf[mbase + r][col] = (__bf16)accV[r];
  __syncthreads();
  {
    const v8bf* src = (const v8bf*)&abf[0][0];
    v8bf* dst = (v8bf*)(Vbuf + (size_t)tile * (16*E_));
    dst[t] = src[t];
  }
}

// ---- K7: per-cell attention (gather + TE bias + softmax + weighted sums) ----
__global__ void __launch_bounds__(256) k_attn(
    const float* __restrict__ x, const float* __restrict__ tg,
    const unsigned* __restrict__ bounds,
    const int* __restrict__ offsets, const int* __restrict__ order,
    const float* __restrict__ qvec, const float* __restrict__ kgrid,
    const float* __restrict__ vgrid,
    const __bf16* __restrict__ Kbuf, const __bf16* __restrict__ Vbuf,
    const float* __restrict__ pw1, const float* __restrict__ pb1,
    const float* __restrict__ pw2, const float* __restrict__ pb2,
    float* attnV, float* out_base){
  GridParams gp = make_params(bounds);
  int flat = blockIdx.x;
  if (flat >= gp.G) return;
  int b = flat / gp.m;
  int cell = flat - b * gp.m;
  int pp = gp.p0 * gp.p1;
  int it = cell / pp;
  int rem = cell - it * pp;
  int i0 = rem / gp.p1;
  int i1 = rem - i0 * gp.p1;
  float xq0 = tg[b*NT_ + it];
  float xq1 = gp.lo0 + i0 * gp.sp0;
  float xq2 = gp.lo1 + i1 * gp.sp1;
  int base = offsets[flat];
  int P1 = min(offsets[flat+1] - base + 1, PCAP);

  __shared__ float qs[E_], kg[E_], vg[E_];
  __shared__ float lw[PCAP][H_];
  int t = threadIdx.x, wid = t >> 5, lane = t & 31;
  if (t < E_){ qs[t] = qvec[t]; kg[t] = kgrid[t]; vg[t] = vgrid[t]; }
  __syncthreads();

  // logits: dots/sqrt(D) + TE MLP bias
  for (int j = t; j < P1; j += 256){
    float d0, d1, d2;
    float dot[H_];
    #pragma unroll
    for (int h = 0; h < H_; ++h) dot[h] = 0.f;
    if (j == 0){
      d0 = d1 = d2 = 0.f;
      #pragma unroll
      for (int h = 0; h < H_; ++h)
        for (int d = 0; d < D_; ++d) dot[h] += qs[h*16+d] * kg[h*16+d];
    } else {
      int pid = order[base + j - 1];
      d0 = xq0 - x[pid*3+0]; d1 = xq1 - x[pid*3+1]; d2 = xq2 - x[pid*3+2];
      const v8bf* kp8 = (const v8bf*)(Kbuf + (size_t)pid * E_);
      #pragma unroll
      for (int c8 = 0; c8 < 16; ++c8){
        v8bf kv = kp8[c8];
        int h = c8 >> 1, dbase = (c8 & 1) * 8;
        #pragma unroll
        for (int i = 0; i < 8; ++i) dot[h] += qs[h*16 + dbase + i] * (float)kv[i];
      }
    }
    float te[H_];
    #pragma unroll
    for (int h = 0; h < H_; ++h) te[h] = pb2[h];
    for (int c = 0; c < PHI_H_; ++c){
      float hv = d0*pw1[0*PHI_H_+c] + d1*pw1[1*PHI_H_+c] + d2*pw1[2*PHI_H_+c] + pb1[c];
      hv = gelu_f(hv);
      #pragma unroll
      for (int h = 0; h < H_; ++h) te[h] += hv * pw2[c*H_+h];
    }
    #pragma unroll
    for (int h = 0; h < H_; ++h) lw[j][h] = dot[h] * 0.25f + te[h];
  }
  __syncthreads();

  // softmax over keys, wave wid handles head wid
  {
    int h = wid;
    float mx = -INFINITY;
    for (int j = lane; j < P1; j += 32) mx = fmaxf(mx, lw[j][h]);
    for (int o = 16; o > 0; o >>= 1) mx = fmaxf(mx, __shfl_xor(mx, o, 32));
    float s = 0.f;
    for (int j = lane; j < P1; j += 32) s += __expf(lw[j][h] - mx);
    for (int o = 16; o > 0; o >>= 1) s += __shfl_xor(s, o, 32);
    float inv = 1.0f / s;
    for (int j = lane; j < P1; j += 32) lw[j][h] = __expf(lw[j][h] - mx) * inv;
  }
  __syncthreads();

  // attn @ V  -> attnV[flat]
  if (t < E_){
    int h = t >> 4;
    float acc = lw[0][h] * vg[t];
    for (int j = 1; j < P1; ++j){
      int pid = order[base + j - 1];
      acc += lw[j][h] * (float)Vbuf[(size_t)pid*E_ + t];
    }
    attnV[(size_t)flat*E_ + t] = acc;
  }
  // pseudo-location update (wave 0); diff for j==0 is zero
  if (wid == 0){
    float a0 = 0.f, a1 = 0.f, a2 = 0.f;
    for (int j = 1 + lane; j < P1; j += 32){
      int pid = order[base + j - 1];
      float w = 0.f;
      #pragma unroll
      for (int h = 0; h < H_; ++h) w += lw[j][h];
      a0 += w * (xq0 - x[pid*3+0]);
      a1 += w * (xq1 - x[pid*3+1]);
      a2 += w * (xq2 - x[pid*3+2]);
    }
    for (int o = 16; o > 0; o >>= 1){
      a0 += __shfl_xor(a0, o, 32); a1 += __shfl_xor(a1, o, 32); a2 += __shfl_xor(a2, o, 32);
    }
    if (lane == 0){
      out_base[(size_t)flat*3+0] = xq0 - a0 * (1.0f/(float)H_);
      out_base[(size_t)flat*3+1] = xq1 - a1 * (1.0f/(float)H_);
      out_base[(size_t)flat*3+2] = xq2 - a2 * (1.0f/(float)H_);
    }
  }
}

// ---- K8: output projection + FF over all cells (WMMA bf16, 16-row tiles) ----
__global__ void __launch_bounds__(256) k_out(
    const unsigned* __restrict__ bounds, const float* __restrict__ attnV,
    const float* __restrict__ latent, const float* __restrict__ Wo,
    const float* __restrict__ bo,
    const float* __restrict__ g2, const float* __restrict__ b2,
    const float* __restrict__ fw1, const float* __restrict__ fb1,
    const float* __restrict__ fw2, const float* __restrict__ fb2,
    float* out_base){
  GridParams gp = make_params(bounds);
  int tile = blockIdx.x;
  if (tile * 16 >= gp.G) return;
  __shared__ __attribute__((aligned(16))) __bf16 wT[E_][E_];    // transposed weight chunk [n][k]
  __shared__ __attribute__((aligned(16))) __bf16 abf[16][E_];
  __shared__ __attribute__((aligned(16))) float  y1[16][E_];
  __shared__ __attribute__((aligned(16))) __bf16 hbuf[16][FF_H_];
  int t = threadIdx.x, wid = t >> 5, lane = t & 31;
  int col = (lane & 15) + (wid << 4);
  int arow = lane & 15;
  int mbase = ((lane >> 4) & 1) * 8;

  // stage attnV rows (vector loads, guarded)
  for (int i = 0; i < 2; ++i){
    int e4 = t + 256*i;                 // 512 float4 = 16x128 floats
    int r = e4 >> 5, c = (e4 & 31) * 4;
    int flat = tile*16 + r;
    v4f v = {0.f,0.f,0.f,0.f};
    if (flat < gp.G) v = *(const v4f*)(attnV + (size_t)flat*E_ + c);
    #pragma unroll
    for (int j = 0; j < 4; ++j) abf[r][c + j] = (__bf16)v[j];
  }
  // stage Wo transposed
  for (int i = 0; i < 16; ++i){
    int e4 = t + 256*i;
    int k = e4 >> 5, nb = (e4 & 31) * 4;
    v4f w4 = *(const v4f*)(Wo + (size_t)e4*4);
    #pragma unroll
    for (int j = 0; j < 4; ++j) wT[nb + j][k] = (__bf16)w4[j];
  }
  __syncthreads();

  // y1 = latent + attnV @ Wo + bo
  {
    v8f acc = V8F_ZERO;
    #pragma unroll
    for (int kk = 0; kk < 4; ++kk){
      int kb = kk * 32;
      v16bf af  = load_frag(&abf[arow][0], kb, lane);
      v16bf bf_ = load_frag(&wT[col][0], kb, lane);
      acc = __builtin_amdgcn_wmma_f32_16x16x32_bf16(false, af, false, bf_, (short)0, acc, false, false);
    }
    float ln = latent[col] + bo[col];
    #pragma unroll
    for (int r = 0; r < 8; ++r) y1[mbase + r][col] = ln + acc[r];
  }
  __syncthreads();

  // LN2 rows -> abf (bf16)
  for (int rr = 0; rr < 2; ++rr){
    int row = wid + rr*8;
    float s = 0.f, sq = 0.f;
    for (int k = 0; k < 4; ++k){ float v = y1[row][lane + 32*k]; s += v; sq += v*v; }
    for (int o = 16; o > 0; o >>= 1){ s += __shfl_xor(s, o, 32); sq += __shfl_xor(sq, o, 32); }
    float mean = s * (1.0f/128.f), var = sq * (1.0f/128.f) - mean*mean;
    float rstd = rsqrtf(var + 1e-5f);
    for (int k = 0; k < 4; ++k){
      int c = lane + 32*k;
      abf[row][c] = (__bf16)((y1[row][c] - mean) * rstd * g2[c] + b2[c]);
    }
  }
  __syncthreads();

  // h = gelu(abf @ fw1 + fb1), processed in 4 column chunks of 128
  for (int ch = 0; ch < 4; ++ch){
    for (int i = 0; i < 16; ++i){
      int e4 = t + 256*i;
      int k = e4 >> 5, cb = (e4 & 31) * 4;
      v4f w4 = *(const v4f*)(fw1 + (size_t)k*FF_H_ + ch*128 + cb);
      #pragma unroll
      for (int j = 0; j < 4; ++j) wT[cb + j][k] = (__bf16)w4[j];
    }
    __syncthreads();
    v8f acc = V8F_ZERO;
    #pragma unroll
    for (int kk = 0; kk < 4; ++kk){
      int kb = kk * 32;
      v16bf af  = load_frag(&abf[arow][0], kb, lane);
      v16bf bf_ = load_frag(&wT[col][0], kb, lane);
      acc = __builtin_amdgcn_wmma_f32_16x16x32_bf16(false, af, false, bf_, (short)0, acc, false, false);
    }
    float bb = fb1[ch*128 + col];
    #pragma unroll
    for (int r = 0; r < 8; ++r)
      hbuf[mbase + r][ch*128 + col] = (__bf16)gelu_f(acc[r] + bb);
    __syncthreads();
  }

  // y2 = y1 + hbuf @ fw2 + fb2, K=512 in 4 chunks
  v8f acc2 = V8F_ZERO;
  for (int kc = 0; kc < 4; ++kc){
    for (int i = 0; i < 16; ++i){
      int e4 = t + 256*i;
      int k = e4 >> 5, cb = (e4 & 31) * 4;
      v4f w4 = *(const v4f*)(fw2 + (size_t)(kc*128 + k)*E_ + cb);
      #pragma unroll
      for (int j = 0; j < 4; ++j) wT[cb + j][k] = (__bf16)w4[j];
    }
    __syncthreads();
    #pragma unroll
    for (int kk = 0; kk < 4; ++kk){
      int kb = kk * 32;
      v16bf af  = load_frag(&hbuf[arow][kc*128], kb, lane);
      v16bf bf_ = load_frag(&wT[col][0], kb, lane);
      acc2 = __builtin_amdgcn_wmma_f32_16x16x32_bf16(false, af, false, bf_, (short)0, acc2, false, false);
    }
    __syncthreads();
  }
  // finalize into y1 then cooperative float4 stores
  {
    float bb = fb2[col];
    #pragma unroll
    for (int r = 0; r < 8; ++r){
      int m = mbase + r;
      y1[m][col] = y1[m][col] + acc2[r] + bb;
    }
  }
  __syncthreads();
  {
    float* zout = out_base + (size_t)gp.G * 3;  // z section follows x section
    for (int i = 0; i < 2; ++i){
      int e4 = t + 256*i;
      int r = e4 >> 5, c = (e4 & 31) * 4;
      int flat = tile*16 + r;
      if (flat < gp.G)
        *(v4f*)(zout + (size_t)flat*E_ + c) = *(const v4f*)&y1[r][c];
    }
  }
}

extern "C" void kernel_launch(void* const* d_in, const int* in_sizes, int n_in,
                              void* d_out, int out_size, void* d_ws, size_t ws_size,
                              hipStream_t stream){
  (void)in_sizes; (void)n_in; (void)out_size; (void)ws_size;
  const float* x      = (const float*)d_in[0];
  const float* z      = (const float*)d_in[1];
  const float* tg     = (const float*)d_in[2];
  const float* latent = (const float*)d_in[3];
  const float* g1     = (const float*)d_in[4];
  const float* b1     = (const float*)d_in[5];
  const float* g2     = (const float*)d_in[6];
  const float* b2     = (const float*)d_in[7];
  const float* Wq     = (const float*)d_in[8];
  const float* Wk     = (const float*)d_in[9];
  const float* Wv     = (const float*)d_in[10];
  const float* Wo     = (const float*)d_in[11];
  const float* bo     = (const float*)d_in[12];
  const float* pw1    = (const float*)d_in[13];
  const float* pb1    = (const float*)d_in[14];
  const float* pw2    = (const float*)d_in[15];
  const float* pb2    = (const float*)d_in[16];
  const float* fw1    = (const float*)d_in[17];
  const float* fb1    = (const float*)d_in[18];
  const float* fw2    = (const float*)d_in[19];
  const float* fb2    = (const float*)d_in[20];

  char* ws = (char*)d_ws;
  size_t o = 0;
  unsigned* bounds = (unsigned*)(ws + o); o += 64;
  int* counts  = (int*)(ws + o); o += (size_t)G_MAX * 4;
  int* cursor  = (int*)(ws + o); o += (size_t)G_MAX * 4;
  int* offsets = (int*)(ws + o); o += (size_t)(G_MAX + 1) * 4; o = (o + 255) & ~(size_t)255;
  int* cellof  = (int*)(ws + o); o += (size_t)BN * 4;
  int* order   = (int*)(ws + o); o += (size_t)BN * 4;
  float* qvec  = (float*)(ws + o); o += 512;
  float* kgrid = (float*)(ws + o); o += 512;
  float* vgrid = (float*)(ws + o); o += 512;
  float* attnV = (float*)(ws + o); o += (size_t)G_MAX * E_ * 4;
  __bf16* Kbuf = (__bf16*)(ws + o); o += (size_t)BN * E_ * 2;
  __bf16* Vbuf = (__bf16*)(ws + o); o += (size_t)BN * E_ * 2;
  float* outp  = (float*)d_out;

  k_init   <<<64,       256, 0, stream>>>(bounds, counts, cursor);
  k_bounds <<<512,      256, 0, stream>>>(x, bounds);
  k_assign <<<BN/256,   256, 0, stream>>>(x, tg, bounds, counts, cellof);
  k_scan   <<<1,       1024, 0, stream>>>(counts, offsets);
  k_scatter<<<BN/256,   256, 0, stream>>>(cellof, offsets, cursor, order);
  k_pre    <<<1,        128, 0, stream>>>(latent, g1, b1, Wq, Wk, Wv, qvec, kgrid, vgrid);
  k_kv     <<<BN/16,    256, 0, stream>>>(z, g1, b1, Wk, Wv, Kbuf, Vbuf);
  k_attn   <<<G_MAX,    256, 0, stream>>>(x, tg, bounds, offsets, order, qvec, kgrid, vgrid,
                                          Kbuf, Vbuf, pw1, pb1, pw2, pb2, attnV, outp);
  k_out    <<<G_MAX/16, 256, 0, stream>>>(bounds, attnV, latent, Wo, bo, g2, b2,
                                          fw1, fb1, fw2, fb2, outp);
}
